// DPSA_31198642438215
// MI455X (gfx1250) — compile-verified, matrix operand order checked by reading, not run on hardware
//
#include <hip/hip_runtime.h>

// ---------------------------------------------------------------------------
// DPSA on MI455X (gfx1250): bf16 WMMA pipeline
//   1) ln_kernel       : LayerNorm over C, write xn bf16 [pos, 256]
//   2) cvt_kernel      : w_qkv / w_out fp32 -> bf16
//   3) gemm_qkv_kernel : xn(73728x256) x Wqkv^T(256x768) -> qkv bf16
//                        (WMMA, 4 N-tiles/wave for A-fragment reuse)
//   4) normalize_kernel: l2-normalize q,k per (pos, head) in place
//   5) qprobe_kernel   : q_probe[b,n,32] = sum over positions
//   6) select_gather   : top-16 rows / cols, gather kf,vf [b,n,256,32]
//   7) attn_kernel     : async-to-LDS K stage, WMMA QK^T (K=32, 1 wmma/tile)
//                        -> softmax -> WMMA PV
//   8) gemm_out_kernel : o(73728x256) x Wout^T(256x256) + bias -> fp32 NCHW
// ---------------------------------------------------------------------------

typedef __attribute__((ext_vector_type(16))) __bf16 v16bf;
typedef __attribute__((ext_vector_type(8)))  __bf16 v8bf;
typedef __attribute__((ext_vector_type(8)))  float  v8f;
typedef __attribute__((ext_vector_type(4)))  int    v4i;

#define N_HEADS 8
#define DH      32
#define HW      9216        // 96*96
#define NPOS    73728       // 8*9216
#define CIN     256
#define OQKV    768
#define NJ      256         // TOP_H*TOP_W

#if defined(__gfx1250__) && __has_builtin(__builtin_amdgcn_global_load_async_to_lds_b128)
#define HAVE_ASYNC_LDS 1
typedef v4i __attribute__((address_space(1)))* gas1_t;   // global int4*
typedef v4i __attribute__((address_space(3)))* las3_t;   // LDS int4*
#else
#define HAVE_ASYNC_LDS 0
#endif

static __device__ __forceinline__ void wait_async0() {
#if __has_builtin(__builtin_amdgcn_s_wait_asynccnt)
  __builtin_amdgcn_s_wait_asynccnt(0);
#else
  asm volatile("s_wait_asynccnt 0x0" ::: "memory");
#endif
}

static __device__ __forceinline__ unsigned short f2bf(float f) {
  unsigned u = __float_as_uint(f);
  u += 0x7FFFu + ((u >> 16) & 1u);          // round-to-nearest-even
  return (unsigned short)(u >> 16);
}
static __device__ __forceinline__ float bf2f(unsigned short h) {
  return __uint_as_float(((unsigned)h) << 16);
}

// Load a 16-bit A/B WMMA fragment per CDNA5 ISA 7.12.2.
// p points at (row base + k0); rows are (ld-elem)-strided, 16B aligned.
// kh = lane>>4 selects the K-half: elems 0..7 = K kh*8..kh*8+7,
// elems 8..15 = K 16+kh*8 .. 16+kh*8+7.
static __device__ __forceinline__ v16bf frag_ld(const __bf16* p, int kh) {
  v8bf lo = *(const v8bf*)(p + kh * 8);
  v8bf hi = *(const v8bf*)(p + 16 + kh * 8);
  v16bf r;
#pragma unroll
  for (int j = 0; j < 8; ++j) { r[j] = lo[j]; r[8 + j] = hi[j]; }
  return r;
}

// ---------------------------------------------------------------- LayerNorm
__global__ void ln_kernel(const float* __restrict__ x,
                          const float* __restrict__ g,
                          const float* __restrict__ bta,
                          unsigned short* __restrict__ xn) {
  int p = blockIdx.x * blockDim.x + threadIdx.x;   // 0..NPOS-1
  if (p >= NPOS) return;
  int b = p / HW, hw = p % HW;
  const float* xp = x + (size_t)b * CIN * HW + hw;
  float s = 0.f, ss = 0.f;
  for (int c = 0; c < CIN; ++c) {
    float v = xp[(size_t)c * HW];
    s += v; ss += v * v;
  }
  float mean = s * (1.f / CIN);
  float var  = ss * (1.f / CIN) - mean * mean;
  float inv  = rsqrtf(var + 1e-5f);
  unsigned short* o = xn + (size_t)p * CIN;
  for (int c = 0; c < CIN; ++c) {
    float v = (xp[(size_t)c * HW] - mean) * inv * g[c] + bta[c];
    o[c] = f2bf(v);
  }
}

// ---------------------------------------------------------- weight convert
__global__ void cvt_kernel(const float* __restrict__ wq,
                           const float* __restrict__ wo,
                           unsigned short* __restrict__ wqb,
                           unsigned short* __restrict__ wob) {
  int i = blockIdx.x * blockDim.x + threadIdx.x;
  if (i < OQKV * CIN) wqb[i] = f2bf(wq[i]);
  if (i < CIN * CIN)  wob[i] = f2bf(wo[i]);
}

// -------------------------------------------------------------- QKV GEMM
// C[M=73728, N=768] = A[M,256] * B[N,256]^T, bf16 in, bf16 out, f32 acc.
// Each wave: 16x64 output (4 N-tiles), A fragment reused across 4 WMMAs.
__global__ void gemm_qkv_kernel(const unsigned short* __restrict__ xn,
                                const unsigned short* __restrict__ wq,
                                unsigned short* __restrict__ qkv) {
  int wave = threadIdx.x >> 5, lane = threadIdx.x & 31;
  int tile = blockIdx.x * 8 + wave;          // < 4608*12
  int tm = tile / 12, tg = tile % 12;        // tg: group of 4 N-tiles
  int ls = lane & 15, kh = lane >> 4;
  const __bf16* arow = (const __bf16*)xn + (size_t)(tm * 16 + ls) * CIN;
  const __bf16* brow = (const __bf16*)wq + (size_t)(tg * 64 + ls) * CIN;
  v8f c0 = {}, c1 = {}, c2 = {}, c3 = {};
#pragma unroll
  for (int kk = 0; kk < CIN; kk += 32) {
    __builtin_prefetch(arow + kk + 64, 0, 1);
    v16bf a  = frag_ld(arow + kk, kh);
    v16bf b0 = frag_ld(brow + kk, kh);
    v16bf b1 = frag_ld(brow + 16 * CIN + kk, kh);
    v16bf b2 = frag_ld(brow + 32 * CIN + kk, kh);
    v16bf b3 = frag_ld(brow + 48 * CIN + kk, kh);
    c0 = __builtin_amdgcn_wmma_f32_16x16x32_bf16(false, a, false, b0, (short)0, c0, false, false);
    c1 = __builtin_amdgcn_wmma_f32_16x16x32_bf16(false, a, false, b1, (short)0, c1, false, false);
    c2 = __builtin_amdgcn_wmma_f32_16x16x32_bf16(false, a, false, b2, (short)0, c2, false, false);
    c3 = __builtin_amdgcn_wmma_f32_16x16x32_bf16(false, a, false, b3, (short)0, c3, false, false);
  }
#pragma unroll
  for (int r = 0; r < 8; ++r) {
    int m = r + kh * 8;
    unsigned short* orow = qkv + (size_t)(tm * 16 + m) * OQKV + tg * 64 + ls;
    orow[0]  = f2bf(c0[r]);
    orow[16] = f2bf(c1[r]);
    orow[32] = f2bf(c2[r]);
    orow[48] = f2bf(c3[r]);
  }
}

// ----------------------------------------------------- q/k L2 normalization
__global__ void normalize_kernel(unsigned short* __restrict__ qkv) {
  int t = blockIdx.x * blockDim.x + threadIdx.x;   // NPOS * N_HEADS
  if (t >= NPOS * N_HEADS) return;
  int n = t & 7, pos = t >> 3;
  unsigned short* q = qkv + (size_t)pos * OQKV + n * DH;
  unsigned short* k = q + 256;
  float sq = 0.f, sk = 0.f;
  for (int d = 0; d < DH; ++d) {
    float a = bf2f(q[d]); sq += a * a;
    float b = bf2f(k[d]); sk += b * b;
  }
  float rq = 1.f / fmaxf(sqrtf(sq), 1e-12f);
  float rk = 1.f / fmaxf(sqrtf(sk), 1e-12f);
  for (int d = 0; d < DH; ++d) {
    q[d] = f2bf(bf2f(q[d]) * rq);
    k[d] = f2bf(bf2f(k[d]) * rk);
  }
}

// ------------------------------------------------------------------ q_probe
__global__ void qprobe_kernel(const unsigned short* __restrict__ qkv,
                              float* __restrict__ qprobe) {
  __shared__ float red[256];
  int bn = blockIdx.x, b = bn >> 3, n = bn & 7;
  int tid = threadIdx.x, d = tid & 31, chunk = tid >> 5;   // 8 chunks
  float s = 0.f;
  for (int p = chunk; p < HW; p += 8)
    s += bf2f(qkv[(size_t)(b * HW + p) * OQKV + n * DH + d]);
  red[chunk * 32 + d] = s;
  __syncthreads();
  if (chunk == 0) {
    float a = red[d];
    for (int c2 = 1; c2 < 8; ++c2) a += red[c2 * 32 + d];
    qprobe[bn * DH + d] = a;
  }
}

// ---------------------------------------------- top-16 rows/cols + gather
__global__ void select_gather_kernel(const unsigned short* __restrict__ qkv,
                                     const float* __restrict__ qprobe,
                                     unsigned short* __restrict__ kf,
                                     unsigned short* __restrict__ vf) {
  __shared__ float qp[DH];
  __shared__ float sc[96];
  __shared__ int idxh[16];
  __shared__ int idxw[16];
  int bn = blockIdx.x, b = bn >> 3, n = bn & 7;
  int tid = threadIdx.x;                                  // 128 threads
  if (tid < DH) qp[tid] = qprobe[bn * DH + tid];
  __syncthreads();
  // score_h[y] = q_probe . sum_x k[y,x,:]
  if (tid < 96) {
    float a = 0.f;
    for (int x = 0; x < 96; ++x) {
      const unsigned short* kp =
          qkv + (size_t)(b * HW + tid * 96 + x) * OQKV + 256 + n * DH;
      for (int d = 0; d < DH; ++d) a += bf2f(kp[d]) * qp[d];
    }
    sc[tid] = a;
  }
  __syncthreads();
  if (tid == 0) {
    for (int i = 0; i < 16; ++i) {
      int best = 0; float bv = -3e38f;
      for (int y = 0; y < 96; ++y) if (sc[y] > bv) { bv = sc[y]; best = y; }
      idxh[i] = best; sc[best] = -3e38f;
    }
  }
  __syncthreads();
  // score_w[x] over selected rows
  if (tid < 96) {
    float a = 0.f;
    for (int yi = 0; yi < 16; ++yi) {
      int y = idxh[yi];
      const unsigned short* kp =
          qkv + (size_t)(b * HW + y * 96 + tid) * OQKV + 256 + n * DH;
      for (int d = 0; d < DH; ++d) a += bf2f(kp[d]) * qp[d];
    }
    sc[tid] = a;
  }
  __syncthreads();
  if (tid == 0) {
    for (int i = 0; i < 16; ++i) {
      int best = 0; float bv = -3e38f;
      for (int x = 0; x < 96; ++x) if (sc[x] > bv) { bv = sc[x]; best = x; }
      idxw[i] = best; sc[best] = -3e38f;
    }
  }
  __syncthreads();
  // gather kf/vf [bn, 256, 32]
  for (int j = tid; j < NJ; j += 128) {
    int y = idxh[j >> 4], x = idxw[j & 15];
    const unsigned short* kp =
        qkv + (size_t)(b * HW + y * 96 + x) * OQKV + 256 + n * DH;
    const unsigned short* vp = kp + 256;
    unsigned short* ko = kf + ((size_t)bn * NJ + j) * DH;
    unsigned short* vo = vf + ((size_t)bn * NJ + j) * DH;
    for (int d = 0; d < DH; ++d) { ko[d] = kp[d]; vo[d] = vp[d]; }
  }
}

// ----------------------------------------------------------------- Attention
// One wave per 16-query tile. K staged via async-to-LDS (ASYNCcnt path) when
// available. sim: 16 WMMA (K=32 each). softmax in-regs with wave32 shuffles.
// P staged in LDS bf16, PV: 8 k-steps x 2 d-tiles WMMA.
__global__ void attn_kernel(const unsigned short* __restrict__ qkv,
                            const unsigned short* __restrict__ kf,
                            const unsigned short* __restrict__ vf,
                            unsigned short* __restrict__ obuf) {
  __shared__ unsigned short kf_s[NJ * DH];        // 16 KB
  __shared__ unsigned short vfT_s[DH * NJ];       // 16 KB (transposed)
  __shared__ unsigned short P_s[2][16 * NJ];      // 16 KB (2 waves/block)
  int bn = blockIdx.y, b = bn >> 3, n = bn & 7;
  int tid = threadIdx.x;                          // 64 threads = 2 waves
  int wave = tid >> 5, lane = tid & 31;
  const unsigned short* kfg = kf + (size_t)bn * NJ * DH;
  const unsigned short* vfg = vf + (size_t)bn * NJ * DH;
#if HAVE_ASYNC_LDS
  // per-lane async b128 copies: 8 bf16 per op, VGPRs bypassed, ASYNCcnt.
  for (int i = tid; i < (NJ * DH) / 8; i += 64) {
    gas1_t g = (gas1_t)(unsigned long long)(kfg + i * 8);
    las3_t l = (las3_t)(unsigned)(unsigned long long)(&kf_s[i * 8]);
    __builtin_amdgcn_global_load_async_to_lds_b128(g, l, 0, 0);
  }
#else
  for (int i = tid; i < NJ * DH; i += 64) kf_s[i] = kfg[i];
#endif
  for (int i = tid; i < NJ * DH; i += 64) {
    int j = i >> 5, d = i & 31;
    vfT_s[d * NJ + j] = vfg[i];
  }
#if HAVE_ASYNC_LDS
  wait_async0();
#endif
  __syncthreads();

  int tileI = blockIdx.x * 2 + wave;              // 0..575
  int posBase = b * HW + tileI * 16;
  int ls = lane & 15, kh = lane >> 4;

  // Q fragment (K = DH = 32 -> single-shot sim WMMAs)
  const __bf16* qrow = (const __bf16*)qkv + (size_t)(posBase + ls) * OQKV + n * DH;
  v16bf qa = frag_ld(qrow, kh);

  v8f acc[16];
#pragma unroll
  for (int t = 0; t < 16; ++t) {
    const __bf16* krow = (const __bf16*)kf_s + (t * 16 + ls) * DH;
    v16bf kb = frag_ld(krow, kh);
    v8f c = {};
    acc[t] = __builtin_amdgcn_wmma_f32_16x16x32_bf16(false, qa, false, kb,
                                                     (short)0, c, false, false);
  }

  // softmax rows: row r+8*kh lives in this lane's half; reduce over 16 lanes
  unsigned short* P = P_s[wave];
#pragma unroll
  for (int r = 0; r < 8; ++r) {
    float m = -3e38f;
#pragma unroll
    for (int t = 0; t < 16; ++t) m = fmaxf(m, acc[t][r]);
#pragma unroll
    for (int off = 1; off < 16; off <<= 1) m = fmaxf(m, __shfl_xor(m, off));
    float s = 0.f;
#pragma unroll
    for (int t = 0; t < 16; ++t) s += __expf(acc[t][r] - m);
#pragma unroll
    for (int off = 1; off < 16; off <<= 1) s += __shfl_xor(s, off);
    float inv = 1.f / s;
    int rr = r + kh * 8;
#pragma unroll
    for (int t = 0; t < 16; ++t)
      P[rr * NJ + t * 16 + ls] = f2bf(__expf(acc[t][r] - m) * inv);
  }
  __syncthreads();

  // PV: A = P (16 x 256), B = V^T (256 x 32) in two 16-wide d tiles
  v8f oc0 = {}, oc1 = {};
#pragma unroll
  for (int kk = 0; kk < NJ; kk += 32) {
    const __bf16* prow = (const __bf16*)P + ls * NJ + kk;
    v16bf pa = frag_ld(prow, kh);
    const __bf16* v0 = (const __bf16*)vfT_s + (size_t)ls * NJ + kk;
    v16bf vb0 = frag_ld(v0, kh);
    oc0 = __builtin_amdgcn_wmma_f32_16x16x32_bf16(false, pa, false, vb0,
                                                  (short)0, oc0, false, false);
    const __bf16* v1 = (const __bf16*)vfT_s + (size_t)(16 + ls) * NJ + kk;
    v16bf vb1 = frag_ld(v1, kh);
    oc1 = __builtin_amdgcn_wmma_f32_16x16x32_bf16(false, pa, false, vb1,
                                                  (short)0, oc1, false, false);
  }
#pragma unroll
  for (int r = 0; r < 8; ++r) {
    int m = r + kh * 8;
    size_t orow = (size_t)(posBase + m) * CIN + n * DH;
    obuf[orow + ls]      = f2bf(oc0[r]);
    obuf[orow + 16 + ls] = f2bf(oc1[r]);
  }
}

// ----------------------------------------------------------- output GEMM
// Each wave: 16x64 output (4 N-tiles), A fragment reused across 4 WMMAs.
__global__ void gemm_out_kernel(const unsigned short* __restrict__ obf,
                                const unsigned short* __restrict__ wo,
                                const float* __restrict__ bias,
                                float* __restrict__ out) {
  int wave = threadIdx.x >> 5, lane = threadIdx.x & 31;
  int tile = blockIdx.x * 8 + wave;          // < 4608*4
  int tm = tile / 4, tg = tile % 4;
  int ls = lane & 15, kh = lane >> 4;
  const __bf16* arow = (const __bf16*)obf + (size_t)(tm * 16 + ls) * CIN;
  const __bf16* brow = (const __bf16*)wo  + (size_t)(tg * 64 + ls) * CIN;
  v8f c0 = {}, c1 = {}, c2 = {}, c3 = {};
#pragma unroll
  for (int kk = 0; kk < CIN; kk += 32) {
    __builtin_prefetch(arow + kk + 64, 0, 1);
    v16bf a  = frag_ld(arow + kk, kh);
    v16bf b0 = frag_ld(brow + kk, kh);
    v16bf b1 = frag_ld(brow + 16 * CIN + kk, kh);
    v16bf b2 = frag_ld(brow + 32 * CIN + kk, kh);
    v16bf b3 = frag_ld(brow + 48 * CIN + kk, kh);
    c0 = __builtin_amdgcn_wmma_f32_16x16x32_bf16(false, a, false, b0, (short)0, c0, false, false);
    c1 = __builtin_amdgcn_wmma_f32_16x16x32_bf16(false, a, false, b1, (short)0, c1, false, false);
    c2 = __builtin_amdgcn_wmma_f32_16x16x32_bf16(false, a, false, b2, (short)0, c2, false, false);
    c3 = __builtin_amdgcn_wmma_f32_16x16x32_bf16(false, a, false, b3, (short)0, c3, false, false);
  }
#pragma unroll
  for (int r = 0; r < 8; ++r) {
    int m = r + kh * 8;
    int pos = tm * 16 + m;
    int bb = pos / HW, hw = pos % HW;
    float* orow = out + (size_t)bb * CIN * HW + hw;
    int o = tg * 64 + ls;
    orow[(size_t)(o)      * HW] = c0[r] + bias[o];
    orow[(size_t)(o + 16) * HW] = c1[r] + bias[o + 16];
    orow[(size_t)(o + 32) * HW] = c2[r] + bias[o + 32];
    orow[(size_t)(o + 48) * HW] = c3[r] + bias[o + 48];
  }
}

// ------------------------------------------------------------------ launcher
extern "C" void kernel_launch(void* const* d_in, const int* in_sizes, int n_in,
                              void* d_out, int out_size, void* d_ws, size_t ws_size,
                              hipStream_t stream) {
  const float* x     = (const float*)d_in[0];
  const float* ln_g  = (const float*)d_in[1];
  const float* ln_b  = (const float*)d_in[2];
  const float* w_qkv = (const float*)d_in[3];
  const float* w_out = (const float*)d_in[4];
  const float* b_out = (const float*)d_in[5];
  float* out = (float*)d_out;

  char* ws = (char*)d_ws;
  size_t off = 0;
  auto alloc = [&](size_t bytes) {
    size_t o = off;
    off = (off + bytes + 255) & ~(size_t)255;
    return o;
  };
  // xn buffer is consumed by gemm_qkv before attn writes obf -> reuse it.
  size_t xn_off  = alloc((size_t)NPOS * CIN * 2);     // xn bf16 / later obf
  size_t qkv_off = alloc((size_t)NPOS * OQKV * 2);    // qkv bf16
  size_t wqb_off = alloc((size_t)OQKV * CIN * 2);
  size_t wob_off = alloc((size_t)CIN * CIN * 2);
  size_t qp_off  = alloc((size_t)64 * DH * 4);
  size_t kf_off  = alloc((size_t)64 * NJ * DH * 2);
  size_t vf_off  = alloc((size_t)64 * NJ * DH * 2);

  unsigned short* xn  = (unsigned short*)(ws + xn_off);
  unsigned short* qkv = (unsigned short*)(ws + qkv_off);
  unsigned short* wqb = (unsigned short*)(ws + wqb_off);
  unsigned short* wob = (unsigned short*)(ws + wob_off);
  float*          qpb = (float*)(ws + qp_off);
  unsigned short* kfb = (unsigned short*)(ws + kf_off);
  unsigned short* vfb = (unsigned short*)(ws + vf_off);
  unsigned short* obf = xn;   // reuse

  ln_kernel<<<NPOS / 256, 256, 0, stream>>>(x, ln_g, ln_b, xn);
  cvt_kernel<<<(OQKV * CIN + 255) / 256, 256, 0, stream>>>(w_qkv, w_out, wqb, wob);
  gemm_qkv_kernel<<<(4608 * 12) / 8, 256, 0, stream>>>(xn, wqb, qkv);
  normalize_kernel<<<(NPOS * N_HEADS) / 256, 256, 0, stream>>>(qkv);
  qprobe_kernel<<<64, 256, 0, stream>>>(qkv, qpb);
  select_gather_kernel<<<64, 128, 0, stream>>>(qkv, qpb, kfb, vfb);
  attn_kernel<<<dim3(288, 64), 64, 0, stream>>>(qkv, kfb, vfb, obf);
  gemm_out_kernel<<<(4608 * 4) / 8, 256, 0, stream>>>(obf, wob, b_out, out);
}
